// SSM_44667659878985
// MI455X (gfx1250) — matrix operand order repaired and checked by reading, not compiled, not run on hardware
//
#include <hip/hip_runtime.h>
#include <hip/hip_bf16.h>

typedef __attribute__((ext_vector_type(16))) __bf16 v16bf;
typedef __attribute__((ext_vector_type(8)))  float  v8f;
typedef __attribute__((ext_vector_type(4)))  unsigned int u32x4;
typedef __attribute__((ext_vector_type(8)))  int    i32x8;
typedef __attribute__((ext_vector_type(4)))  int    i32x4;

#define A_STRIDE 260   // 256 + 4 DWORD pad -> conflict-avoiding LDS row stride

__device__ __forceinline__ unsigned short f2bf(float f) {
    unsigned int u = __float_as_uint(f);
    u += 0x7FFFu + ((u >> 16) & 1u);   // round-to-nearest-even
    return (unsigned short)(u >> 16);
}

// ---- TDM: DMA a 256x256 f32 matrix from global into LDS with 4-DWORD row padding ----
// D# group0: count=1 | lds_addr | global_addr(57b) | type=2
// D# group1: data_size=4B, pad_enable, pad_interval=256 DWORDs (code 7), pad_amount=4 DWORDs
//            (code 3), tensor 256x256, tile 256x256, dim0_stride=256
__device__ __forceinline__ void tdm_load_mat256(const float* gptr, unsigned ldsByteOff) {
#if __has_builtin(__builtin_amdgcn_tensor_load_to_lds)
    unsigned long long ga = (unsigned long long)gptr;
    u32x4 g0 = { 1u, ldsByteOff, (unsigned)ga, (unsigned)(ga >> 32) | 0x80000000u };
    i32x8 g1 = { (int)((2u << 16) | (1u << 20) | (7u << 22) | (3u << 25)),
                 (int)(256u << 16),   // tensor_dim0 = 256 (bits 79:48)
                 (int)(256u << 16),   // tensor_dim1 = 256 (bits 111:80)
                 (int)(256u << 16),   // tile_dim0   = 256 (bits 127:112)
                 256,                 // tile_dim1   = 256
                 256,                 // tensor_dim0_stride = 256
                 0, 0 };
    i32x4 gz = {0, 0, 0, 0};
#if defined(__clang_major__) && (__clang_major__ >= 23)
    i32x8 gz8 = {0, 0, 0, 0, 0, 0, 0, 0};
    __builtin_amdgcn_tensor_load_to_lds(g0, g1, gz, gz, gz8, 0);
#else
    __builtin_amdgcn_tensor_load_to_lds(g0, g1, gz, gz, 0);
#endif
    __builtin_amdgcn_s_wait_tensorcnt(0);
#else
    (void)gptr; (void)ldsByteOff; // fallback handled by caller
#endif
}

// ---------------- f32 -> bf16 conversion, 8 elems/thread ----------------
__global__ __launch_bounds__(256) void cvt_bf16_kernel(const float* __restrict__ src,
                                                       unsigned short* __restrict__ dst, int n) {
    int i = (blockIdx.x * 256 + threadIdx.x) * 8;
    if (i >= n) return;
    float4 a = *(const float4*)(src + i);
    float4 b = *(const float4*)(src + i + 4);
    union { unsigned short h[8]; uint4 q; } o;
    o.h[0] = f2bf(a.x); o.h[1] = f2bf(a.y); o.h[2] = f2bf(a.z); o.h[3] = f2bf(a.w);
    o.h[4] = f2bf(b.x); o.h[5] = f2bf(b.y); o.h[6] = f2bf(b.z); o.h[7] = f2bf(b.w);
    *(uint4*)(dst + i) = o.q;
}

// ---------------- GEMM1: xb[n,s] = sum_d x[n,d] * Bmat[s,d]  (bf16 WMMA) --------
__global__ __launch_bounds__(512) void gemm1_kernel(const unsigned short* __restrict__ xbf,
                                                    const unsigned short* __restrict__ bmbf,
                                                    float* __restrict__ xb) {
    const int lane = threadIdx.x & 31;
    const int wave = threadIdx.x >> 5;    // 0..15 -> 16-wide S column tile
    const int g    = lane >> 4;
    const int m    = lane & 15;
    const int rowTile = blockIdx.x;       // 0..1023
    const unsigned short* aRow = xbf  + (size_t)(rowTile * 16 + m) * 1024;
    const unsigned short* bRow = bmbf + (size_t)(wave * 16 + m) * 1024;
    v8f acc = {};
    for (int k0 = 0; k0 < 1024; k0 += 32) {
        union { v16bf v; uint4 q[2]; } A, B;
        A.q[0] = *(const uint4*)(aRow + k0 + g * 8);
        A.q[1] = *(const uint4*)(aRow + k0 + g * 8 + 16);
        B.q[0] = *(const uint4*)(bRow + k0 + g * 8);
        B.q[1] = *(const uint4*)(bRow + k0 + g * 8 + 16);
        acc = __builtin_amdgcn_wmma_f32_16x16x32_bf16(false, A.v, false, B.v,
                                                      (short)0, acc, false, false);
    }
    float* out = xb + (size_t)(rowTile * 16 + g * 8) * 256 + wave * 16 + m;
#pragma unroll
    for (int v = 0; v < 8; ++v) out[(size_t)v * 256] = acc[v];
}

// ---------------- small fp32 matrix square: out = in * in  (256x256) ----------------
__global__ __launch_bounds__(256) void matsq_kernel(const float* __restrict__ in,
                                                    float* __restrict__ out) {
    int i = blockIdx.x, j = threadIdx.x;
    float acc = 0.f;
    for (int k = 0; k < 256; ++k) acc += in[i * 256 + k] * in[k * 256 + j];
    out[i * 256 + j] = acc;
}

// ---------------- chunked scan: h_t = A h_{t-1} + xb_t, T=64, split-K over 2 halves ----
// mode 0: start from zero, write only chunk end state E[b,c][:]
// mode 1: start from carry[b,c], write every state as bf16
__global__ __launch_bounds__(512) void scan_chunk_kernel(const float* __restrict__ A,
                                                         const float* __restrict__ xb,
                                                         const float* __restrict__ carry,
                                                         float* __restrict__ E,
                                                         unsigned short* __restrict__ statesbf,
                                                         int mode) {
    extern __shared__ float smem[];
    float* Ar   = smem;                        // A rows, stride A_STRIDE (TDM-padded)
    float* st   = smem + 256 * A_STRIDE;       // shared state [256]
    float* part = st + 256;                    // split-K partials [512]
    const int s    = threadIdx.x & 255;
    const int half = threadIdx.x >> 8;         // 0/1 : K range [half*128, half*128+128)
    const int c = blockIdx.x, b = blockIdx.y;

#if __has_builtin(__builtin_amdgcn_tensor_load_to_lds)
    if (threadIdx.x < 32) {
        tdm_load_mat256(A, (unsigned)__builtin_amdgcn_groupstaticsize());
    }
#else
    for (int r = half; r < 256; r += 2)
        for (int k = s; k < 256; k += 256) Ar[(size_t)r * A_STRIDE + k] = A[(size_t)r * 256 + k];
#endif
    if (half == 0)
        st[s] = (mode == 0) ? 0.f : carry[((size_t)b * 32 + c) * 256 + s];
    __syncthreads();

    const float4* arow4 = (const float4*)(Ar + (size_t)s * A_STRIDE) + half * 32;
    const float4* st4   = (const float4*)st + half * 32;
    const size_t n0 = (size_t)b * 2048 + (size_t)c * 64;
    for (int t = 0; t < 64; ++t) {
        float acc = 0.f;
#pragma unroll 8
        for (int kk = 0; kk < 32; ++kk) {     // 128 K values per half, float4 LDS reads
            float4 a = arow4[kk];
            float4 v = st4[kk];
            acc += a.x * v.x + a.y * v.y + a.z * v.z + a.w * v.w;
        }
        part[threadIdx.x] = acc;
        __syncthreads();
        if (half == 0) {
            float nv = part[s] + part[s + 256] + xb[(n0 + t) * 256 + s];
            st[s] = nv;
            if (mode == 1) statesbf[(n0 + t) * 256 + s] = f2bf(nv);
        }
        __syncthreads();
    }
    if (mode == 0 && half == 0) E[((size_t)b * 32 + c) * 256 + s] = st[s];
}

// ---------------- carry propagation: carry_{c+1} = P carry_c + E_c  (P = A^64) ----------------
__global__ __launch_bounds__(256) void carry_scan_kernel(const float* __restrict__ P,
                                                         const float* __restrict__ E,
                                                         float* __restrict__ carry) {
    extern __shared__ float smem[];
    float* Pr = smem;
    float* st = smem + 256 * A_STRIDE;
    const int s = threadIdx.x, b = blockIdx.x;
#if __has_builtin(__builtin_amdgcn_tensor_load_to_lds)
    if (threadIdx.x < 32) {
        tdm_load_mat256(P, (unsigned)__builtin_amdgcn_groupstaticsize());
    }
#else
    for (int r = 0; r < 256; ++r) Pr[(size_t)r * A_STRIDE + s] = P[(size_t)r * 256 + s];
#endif
    st[s] = 0.f;
    __syncthreads();
    const float4* prow4 = (const float4*)(Pr + (size_t)s * A_STRIDE);
    for (int c = 0; c < 32; ++c) {
        carry[((size_t)b * 32 + c) * 256 + s] = st[s];     // incoming state for chunk c
        float acc = E[((size_t)b * 32 + c) * 256 + s];
#pragma unroll 8
        for (int kk = 0; kk < 64; ++kk) {
            float4 a = prow4[kk];
            float4 v = ((const float4*)st)[kk];
            acc += a.x * v.x + a.y * v.y + a.z * v.z + a.w * v.w;
        }
        __syncthreads();
        st[s] = acc;
        __syncthreads();
    }
}

// ---------------- GEMM2 + GELU + LayerNorm, fused over full D=1024 row ----------------
__global__ __launch_bounds__(512) void gemm2_kernel(const unsigned short* __restrict__ sbf,
                                                    const unsigned short* __restrict__ cbf,
                                                    const float* __restrict__ gamma,
                                                    const float* __restrict__ beta,
                                                    float* __restrict__ out) {
    __shared__ float lds[16 * 1024];          // 64 KB GELU(y) slab
    const int lane = threadIdx.x & 31;
    const int wave = threadIdx.x >> 5;        // 0..15, owns 4 col tiles
    const int g = lane >> 4;
    const int m = lane & 15;
    const int rowTile = blockIdx.x;           // 0..1023
    const unsigned short* aRow = sbf + (size_t)(rowTile * 16 + m) * 256;
    v8f acc[4] = {};
    for (int k0 = 0; k0 < 256; k0 += 32) {
        union U { v16bf v; uint4 q[2]; };
        U A;
        A.q[0] = *(const uint4*)(aRow + k0 + g * 8);
        A.q[1] = *(const uint4*)(aRow + k0 + g * 8 + 16);
#pragma unroll
        for (int j = 0; j < 4; ++j) {
            const unsigned short* bRow = cbf + (size_t)(wave * 64 + j * 16 + m) * 256;
            U B;
            B.q[0] = *(const uint4*)(bRow + k0 + g * 8);
            B.q[1] = *(const uint4*)(bRow + k0 + g * 8 + 16);
            acc[j] = __builtin_amdgcn_wmma_f32_16x16x32_bf16(false, A.v, false, B.v,
                                                             (short)0, acc[j], false, false);
        }
    }
#pragma unroll
    for (int j = 0; j < 4; ++j) {
        int col = wave * 64 + j * 16 + m;
#pragma unroll
        for (int v = 0; v < 8; ++v) {
            float x = acc[j][v];
            lds[(v + 8 * g) * 1024 + col] = 0.5f * x * (1.f + erff(x * 0.70710678118f));
        }
    }
    __syncthreads();
    const int r = wave;                       // wave w normalizes row w
    float sum = 0.f, sq = 0.f;
    for (int cc = lane; cc < 1024; cc += 32) {
        float v = lds[r * 1024 + cc];
        sum += v; sq += v * v;
    }
#pragma unroll
    for (int off = 16; off >= 1; off >>= 1) {
        sum += __shfl_xor(sum, off);
        sq  += __shfl_xor(sq, off);
    }
    const float mean = sum * (1.f / 1024.f);
    const float var  = sq * (1.f / 1024.f) - mean * mean;
    const float rstd = rsqrtf(var + 1e-5f);
    float* orow = out + (size_t)(rowTile * 16 + r) * 1024;
    for (int cc = lane; cc < 1024; cc += 32) {
        float v = (lds[r * 1024 + cc] - mean) * rstd;
        orow[cc] = v * gamma[cc] + beta[cc];
    }
}

extern "C" void kernel_launch(void* const* d_in, const int* in_sizes, int n_in,
                              void* d_out, int out_size, void* d_ws, size_t ws_size,
                              hipStream_t stream) {
    (void)in_sizes; (void)n_in; (void)out_size; (void)ws_size;
    const float* x     = (const float*)d_in[0];
    const float* Amat  = (const float*)d_in[1];
    const float* Bmat  = (const float*)d_in[2];
    const float* Cmat  = (const float*)d_in[3];
    const float* gamma = (const float*)d_in[4];
    const float* beta  = (const float*)d_in[5];
    float* y = (float*)d_out;

    const int Bsz = 8, L = 2048, D = 1024, S = 256;
    const size_t N = (size_t)Bsz * L;   // 16384

    char* ws = (char*)d_ws;
    size_t off = 0;
    auto alloc = [&](size_t bytes) -> char* {
        char* p = ws + off; off += (bytes + 255) & ~(size_t)255; return p;
    };
    unsigned short* x_bf  = (unsigned short*)alloc(N * D * 2);
    unsigned short* bm_bf = (unsigned short*)alloc((size_t)S * D * 2);
    unsigned short* c_bf  = (unsigned short*)alloc((size_t)D * S * 2);
    float*          xb    = (float*)alloc(N * S * 4);
    unsigned short* st_bf = (unsigned short*)alloc(N * S * 2);
    float*          E     = (float*)alloc((size_t)Bsz * 32 * S * 4);
    float*          carry = (float*)alloc((size_t)Bsz * 32 * S * 4);
    float*          Pa    = (float*)alloc((size_t)S * S * 4);
    float*          Pb    = (float*)alloc((size_t)S * S * 4);

    {
        int n = (int)(N * D);
        cvt_bf16_kernel<<<n / 2048, 256, 0, stream>>>(x, x_bf, n);
        n = S * D;
        cvt_bf16_kernel<<<n / 2048, 256, 0, stream>>>(Bmat, bm_bf, n);
        cvt_bf16_kernel<<<n / 2048, 256, 0, stream>>>(Cmat, c_bf, n);
    }

    gemm1_kernel<<<1024, 512, 0, stream>>>(x_bf, bm_bf, xb);

    matsq_kernel<<<256, 256, 0, stream>>>(Amat, Pa);   // A^2
    matsq_kernel<<<256, 256, 0, stream>>>(Pa, Pb);     // A^4
    matsq_kernel<<<256, 256, 0, stream>>>(Pb, Pa);     // A^8
    matsq_kernel<<<256, 256, 0, stream>>>(Pa, Pb);     // A^16
    matsq_kernel<<<256, 256, 0, stream>>>(Pb, Pa);     // A^32
    matsq_kernel<<<256, 256, 0, stream>>>(Pa, Pb);     // A^64

    const size_t scanShmem  = (size_t)(256 * A_STRIDE + 256 + 512) * sizeof(float);
    const size_t carryShmem = (size_t)(256 * A_STRIDE + 256) * sizeof(float);
    hipFuncSetAttribute(reinterpret_cast<const void*>(scan_chunk_kernel),
                        hipFuncAttributeMaxDynamicSharedMemorySize, (int)scanShmem);
    hipFuncSetAttribute(reinterpret_cast<const void*>(carry_scan_kernel),
                        hipFuncAttributeMaxDynamicSharedMemorySize, (int)carryShmem);

    scan_chunk_kernel<<<dim3(32, 8), 512, scanShmem, stream>>>(Amat, xb, nullptr, E, nullptr, 0);
    carry_scan_kernel<<<8, 256, carryShmem, stream>>>(Pb, E, carry);
    scan_chunk_kernel<<<dim3(32, 8), 512, scanShmem, stream>>>(Amat, xb, carry, nullptr, st_bf, 1);

    gemm2_kernel<<<1024, 512, 0, stream>>>(st_bf, c_bf, gamma, beta, y);
}
